// FlowHead_54597624266886
// MI455X (gfx1250) — matrix operand order repaired and checked by reading, not compiled, not run on hardware
//
#include <hip/hip_runtime.h>

// ---------------------------------------------------------------------------
// FlowHead on MI455X (gfx1250, wave32, WMMA).
//
// Roofline: conv1 = 87 GFLOP (implicit GEMM M=256, N=147456, K=1152),
// conv2 = 1.4 GFLOP.  Total HBM traffic ~380 MB -> ~16us @ 23.3 TB/s, so
// conv1 is matrix-core bound.  We use the bf16 WMMA pipe (K=32/instr, ~8x the
// f32 WMMA rate) with a 3-term hi/lo bf16 split for ~f32 accuracy at ~2.7x
// fewer matrix instructions than V_WMMA_F32_16X16X4_F32 would need.
// conv2 has M=2 -> WMMA would be 87% idle; it runs on VALU fmaf.
// ---------------------------------------------------------------------------

typedef __attribute__((ext_vector_type(16))) __bf16 v16bf;
typedef __attribute__((ext_vector_type(8)))  __bf16 v8bf;
typedef __attribute__((ext_vector_type(8)))  float  v8f;

#define B_    16
#define CIN_  128
#define CH_   256
#define H_    96
#define W_    96
#define K1_   (CIN_ * 9)      // 1152 (K ordering: (kh*3+kw)*128 + c)
#define COLS_ 34              // 32-wide output strip + 2 halo
#define CP_   136             // channel stride in LDS (128 + 8 pad -> 16B-aligned b128)

#define WMMA_BF16(A, Bv, C) \
  __builtin_amdgcn_wmma_f32_16x16x32_bf16(false, (A), false, (Bv), (short)0, (C), false, false)

static __device__ inline v16bf cat8(v8bf a, v8bf b) {
  return __builtin_shufflevector(a, b, 0,1,2,3,4,5,6,7,8,9,10,11,12,13,14,15);
}

// A-matrix (16x32 bf16) per-lane VGPR layout (ISA 7.12.2):
// lanes 0-15 hold K groups {0..7},{16..23}; lanes 16-31 hold {8..15},{24..31}.
// Both 8-element groups are contiguous in K -> two global_load_b128.
static __device__ inline v16bf load_a(const __bf16* __restrict__ wp, int row,
                                      int kbase, int half) {
  const __bf16* p = wp + (size_t)row * K1_ + kbase + half * 8;
  return cat8(*(const v8bf*)p, *(const v8bf*)(p + 16));
}

// B-matrix (32x16 bf16): lane n holds K = half*8 + {0..7} and 16 + half*8 + {0..7}
// (same order as A) in the channel-fastest LDS patch -> two 16B-aligned ds_load_b128.
static __device__ inline v16bf load_b(const __bf16* s, int kh, int col,
                                      int cb, int half) {
  const __bf16* p = s + (kh * COLS_ + col) * CP_ + cb + half * 8;
  return cat8(*(const v8bf*)p, *(const v8bf*)(p + 16));
}

// --- prep: permute w1 [O][C][3][3] -> [O][(kh*3+kw)*128+c] and split hi/lo ---
__global__ void split_w1_kernel(const float* __restrict__ w1,
                                unsigned short* __restrict__ whr,
                                unsigned short* __restrict__ wlr) {
  int i = blockIdx.x * 256 + threadIdx.x;
  if (i >= CH_ * K1_) return;
  int o = i / K1_, k = i % K1_;
  int t = k / CIN_, c = k % CIN_;
  int kh = t / 3, kw = t % 3;
  float v = w1[(((size_t)o * CIN_ + c) * 3 + kh) * 3 + kw];
  __bf16 hi = (__bf16)v;
  __bf16 lo = (__bf16)(v - (float)hi);
  ((__bf16*)whr)[i] = hi;
  ((__bf16*)wlr)[i] = lo;
}

// --- conv1 + bias + relu: implicit GEMM, split-bf16 WMMA ---------------------
__global__ __launch_bounds__(256) void conv1_wmma(
    const float* __restrict__ x, const unsigned short* __restrict__ whr,
    const unsigned short* __restrict__ wlr, const float* __restrict__ b1,
    float* __restrict__ hout) {
  const __bf16* wh = (const __bf16*)whr;
  const __bf16* wl = (const __bf16*)wlr;

  __shared__ __bf16 s_hi[3 * COLS_ * CP_];   // 27.7 KB
  __shared__ __bf16 s_lo[3 * COLS_ * CP_];   // 27.7 KB (320 KB/WGP available)

  const int tid = threadIdx.x;
  const int w0  = blockIdx.x * 32;   // 32-wide output strip
  const int y   = blockIdx.y;
  const int bb  = blockIdx.z;

  // Cooperative fill: split f32 -> bf16 hi/lo ONCE per element (shared by all
  // 16 M-tiles).  Consecutive tids walk consecutive w (coalesced NCHW reads).
  for (int id = tid; id < CIN_ * 3 * COLS_; id += 256) {
    int col = id % COLS_;
    int t   = id / COLS_;
    int row = t % 3;
    int c   = t / 3;
    int gy = y + row - 1;
    int gx = w0 + col - 1;
    float v = 0.f;
    if ((unsigned)gy < H_ && (unsigned)gx < W_)
      v = x[(((size_t)bb * CIN_ + c) * H_ + gy) * W_ + gx];
    __bf16 hi = (__bf16)v;
    __bf16 lo = (__bf16)(v - (float)hi);
    int a = (row * COLS_ + col) * CP_ + c;
    s_hi[a] = hi;
    s_lo[a] = lo;
  }
  __syncthreads();

  const int wv = tid >> 5, lane = tid & 31;
  const int half = lane >> 4, lm = lane & 15;
  const int m0 = wv * 32;            // each wave: 2 M-tiles x 2 N-tiles

  v8f acc00 = {}, acc01 = {}, acc10 = {}, acc11 = {};

  for (int kh = 0; kh < 3; ++kh)
    for (int kw = 0; kw < 3; ++kw)
      for (int cb = 0; cb < CIN_; cb += 32) {
        const int kbase = (kh * 3 + kw) * CIN_ + cb;
        if (kbase + 256 < K1_)       // warm next weight block (global_prefetch)
          __builtin_prefetch(wh + (size_t)(m0 + lm) * K1_ + kbase + 256, 0, 3);

        v16bf ah0 = load_a(wh, m0 + lm,      kbase, half);
        v16bf al0 = load_a(wl, m0 + lm,      kbase, half);
        v16bf ah1 = load_a(wh, m0 + 16 + lm, kbase, half);
        v16bf al1 = load_a(wl, m0 + 16 + lm, kbase, half);
        v16bf bh0 = load_b(s_hi, kh, lm + kw,      cb, half);
        v16bf bl0 = load_b(s_lo, kh, lm + kw,      cb, half);
        v16bf bh1 = load_b(s_hi, kh, 16 + lm + kw, cb, half);
        v16bf bl1 = load_b(s_lo, kh, 16 + lm + kw, cb, half);

        // 3-term split product per tile (drop lo*lo; ~2^-16 rel. error)
        acc00 = WMMA_BF16(ah0, bh0, acc00);
        acc00 = WMMA_BF16(ah0, bl0, acc00);
        acc00 = WMMA_BF16(al0, bh0, acc00);
        acc01 = WMMA_BF16(ah0, bh1, acc01);
        acc01 = WMMA_BF16(ah0, bl1, acc01);
        acc01 = WMMA_BF16(al0, bh1, acc01);
        acc10 = WMMA_BF16(ah1, bh0, acc10);
        acc10 = WMMA_BF16(ah1, bl0, acc10);
        acc10 = WMMA_BF16(al1, bh0, acc10);
        acc11 = WMMA_BF16(ah1, bh1, acc11);
        acc11 = WMMA_BF16(ah1, bl1, acc11);
        acc11 = WMMA_BF16(al1, bh1, acc11);
      }

  // Epilogue: C/D layout -> VGPR r holds M = r + 8*half, N = lane&15.
  const int xo0 = w0 + lm, xo1 = w0 + 16 + lm;
#pragma unroll
  for (int r = 0; r < 8; ++r) {
    int o0 = m0 + r + half * 8;
    int o1 = o0 + 16;
    size_t base0 = (((size_t)bb * CH_ + o0) * H_ + y) * W_;
    size_t base1 = (((size_t)bb * CH_ + o1) * H_ + y) * W_;
    float v;
    v = acc00[r] + b1[o0]; hout[base0 + xo0] = v > 0.f ? v : 0.f;
    v = acc01[r] + b1[o0]; hout[base0 + xo1] = v > 0.f ? v : 0.f;
    v = acc10[r] + b1[o1]; hout[base1 + xo0] = v > 0.f ? v : 0.f;
    v = acc11[r] + b1[o1]; hout[base1 + xo1] = v > 0.f ? v : 0.f;
  }
}

// --- conv2 + bias: direct conv on VALU (M=2 -> WMMA would be 87% idle) -------
__global__ __launch_bounds__(256) void conv2_direct(
    const float* __restrict__ hbuf, const float* __restrict__ w2,
    const float* __restrict__ b2, float* __restrict__ out) {
  __shared__ float sw[2 * CH_ * 9];   // all of w2: 18 KB
  __shared__ float st[32 * 18 * 18];  // 32-ch x 18x18 halo tile: 41.5 KB

  const int tid = threadIdx.x;
  const int x0 = blockIdx.x * 16, y0 = blockIdx.y * 16, bb = blockIdx.z;

  for (int id = tid; id < 2 * CH_ * 9; id += 256) sw[id] = w2[id];

  const int lx = tid & 15, ly = tid >> 4;
  float acc0 = 0.f, acc1 = 0.f;

  for (int cb = 0; cb < CH_; cb += 32) {
    __syncthreads();
    for (int id = tid; id < 32 * 18 * 18; id += 256) {
      int cc = id % 18;
      int t  = id / 18;
      int rr = t % 18;
      int c  = t / 18;
      int gy = y0 + rr - 1, gx = x0 + cc - 1;
      float v = 0.f;
      if ((unsigned)gy < H_ && (unsigned)gx < W_)
        v = hbuf[(((size_t)bb * CH_ + cb + c) * H_ + gy) * W_ + gx];
      st[id] = v;
    }
    __syncthreads();
    for (int c = 0; c < 32; ++c) {
      const float* wp0 = &sw[(cb + c) * 9];
      const float* wp1 = &sw[CH_ * 9 + (cb + c) * 9];
#pragma unroll
      for (int kh = 0; kh < 3; ++kh)
#pragma unroll
        for (int kw = 0; kw < 3; ++kw) {
          float v = st[(c * 18 + ly + kh) * 18 + lx + kw];
          acc0 = fmaf(v, wp0[kh * 3 + kw], acc0);
          acc1 = fmaf(v, wp1[kh * 3 + kw], acc1);
        }
    }
  }
  size_t ob = (((size_t)bb * 2) * H_ + (y0 + ly)) * W_ + x0 + lx;
  out[ob] = acc0 + b2[0];
  out[ob + (size_t)H_ * W_] = acc1 + b2[1];
}

// ---------------------------------------------------------------------------
extern "C" void kernel_launch(void* const* d_in, const int* in_sizes, int n_in,
                              void* d_out, int out_size, void* d_ws, size_t ws_size,
                              hipStream_t stream) {
  (void)in_sizes; (void)n_in; (void)out_size; (void)ws_size;
  const float* x  = (const float*)d_in[0];
  const float* w1 = (const float*)d_in[1];
  const float* b1 = (const float*)d_in[2];
  const float* w2 = (const float*)d_in[3];
  const float* b2 = (const float*)d_in[4];
  float* out = (float*)d_out;

  // Workspace: h [16,256,96,96] f32 (151 MB) + split-bf16 w1 (2 x 0.59 MB).
  const size_t hbytes = (size_t)B_ * CH_ * H_ * W_ * sizeof(float);
  float* hbuf = (float*)d_ws;
  unsigned short* wh = (unsigned short*)((char*)d_ws + hbytes);
  unsigned short* wl = wh + (size_t)CH_ * K1_;

  split_w1_kernel<<<(CH_ * K1_ + 255) / 256, 256, 0, stream>>>(w1, wh, wl);
  conv1_wmma<<<dim3(W_ / 32, H_, B_), 256, 0, stream>>>(x, wh, wl, b1, hbuf);
  conv2_direct<<<dim3(W_ / 16, H_ / 16, B_), 256, 0, stream>>>(hbuf, w2, b2, out);
}
